// TA_PECL_Block_72997264163302
// MI455X (gfx1250) — matrix operand order repaired and checked by reading, not compiled, not run on hardware
//
#include <hip/hip_runtime.h>
#include <hip/hip_bf16.h>

typedef __attribute__((ext_vector_type(16))) __bf16 v16bf;
typedef __attribute__((ext_vector_type(8)))  __bf16 v8bf;
typedef __attribute__((ext_vector_type(8)))  float  v8f;

#define Bn 8
#define Sn 2048
#define Dn 2048
#define En 8
#define Rn 16
// pooling chunks
#define PC 32
#define PROWS (Sn / PC)   // 64

// ---------------------------------------------------------------------------
// Kernel 1: partial pooling sums.  grid (PC, B), block 256.
// partial[(b*PC + c)*Dn + d] = sum over 64 rows of X[b, c*64 + s, d]
// ---------------------------------------------------------------------------
__global__ void pool_kernel(const float* __restrict__ X, float* __restrict__ partial) {
    const int c = blockIdx.x, b = blockIdx.y, t = threadIdx.x;
    float acc[8] = {0.f, 0.f, 0.f, 0.f, 0.f, 0.f, 0.f, 0.f};
    const float* base = X + ((size_t)b * Sn + (size_t)c * PROWS) * Dn;
    for (int s = 0; s < PROWS; ++s) {
        const float* row = base + (size_t)s * Dn;
#pragma unroll
        for (int i = 0; i < 8; ++i) acc[i] += row[t + i * 256];
    }
    float* dst = partial + ((size_t)(b * PC + c)) * Dn;
#pragma unroll
    for (int i = 0; i < 8; ++i) dst[t + i * 256] = acc[i];
}

// ---------------------------------------------------------------------------
// Kernel 2: router. 1 block, 256 threads. wave w handles sample b = w.
// ---------------------------------------------------------------------------
__global__ void router_kernel(const float* __restrict__ partial,
                              const float* __restrict__ gw,
                              const float* __restrict__ tb,
                              const float* __restrict__ mb,
                              const int* __restrict__ task_id,
                              const int* __restrict__ mode_id,
                              int* __restrict__ top2i, float* __restrict__ top2w) {
    __shared__ float lg[Bn][En];
    const int tid = threadIdx.x, w = tid >> 5, ln = tid & 31;
    const int b = w;
    // pooled mean held in registers: lane ln owns d = ln + 32*i
    float pooled[64];
#pragma unroll
    for (int i = 0; i < 64; ++i) {
        const int d = ln + i * 32;
        float p = 0.f;
#pragma unroll
        for (int c = 0; c < PC; ++c) p += partial[((size_t)(b * PC + c)) * Dn + d];
        pooled[i] = p * (1.0f / (float)Sn);
    }
    for (int e = 0; e < En; ++e) {
        float acc = 0.f;
        const float* g = gw + (size_t)e * Dn;
#pragma unroll
        for (int i = 0; i < 64; ++i) acc += pooled[i] * g[ln + i * 32];
#pragma unroll
        for (int off = 16; off; off >>= 1) acc += __shfl_xor(acc, off, 32);
        if (ln == 0)
            lg[b][e] = acc + tb[task_id[0] * En + e] + mb[mode_id[0] * En + e];
    }
    __syncthreads();
    if (tid < Bn) {
        const int bb = tid;
        float v[En];
#pragma unroll
        for (int e = 0; e < En; ++e) v[e] = lg[bb][e];
        int i0 = 0;
#pragma unroll
        for (int e = 1; e < En; ++e) if (v[e] > v[i0]) i0 = e;
        int i1 = (i0 == 0) ? 1 : 0;
#pragma unroll
        for (int e = 0; e < En; ++e) if (e != i0 && v[e] > v[i1]) i1 = e;
        const float w0 = 1.0f / (1.0f + __expf(v[i1] - v[i0]));
        top2i[bb * 2 + 0] = i0;  top2i[bb * 2 + 1] = i1;
        top2w[bb * 2 + 0] = w0;  top2w[bb * 2 + 1] = 1.0f - w0;   // scaling = ALPHA/R = 1
    }
}

// -------- WMMA operand helpers (ISA 16-bit A 16x32 / B 32x16 layouts) -------
__device__ __forceinline__ v16bf make16(v8bf lo, v8bf hi) {
    v16bf r;
#pragma unroll
    for (int i = 0; i < 8; ++i) { r[i] = lo[i]; r[i + 8] = hi[i]; }
    return r;
}

// ---------------------------------------------------------------------------
// Kernel 3: mid[b][j][s][r] = top2w[b][j] * (X[b] @ A_e^T)  (bf16 out)
// grid (S/128, 2, B), block 256 (8 waves, 16 rows each).
// ---------------------------------------------------------------------------
#define LPAD 40   // padded LDS row stride in bf16 elements (80B -> conflict-free b128 reads)
__global__ void mid_kernel(const float* __restrict__ X,
                           const float* __restrict__ loraA,
                           const int* __restrict__ top2i,
                           const float* __restrict__ top2w,
                           __bf16* __restrict__ mid) {
    __shared__ __attribute__((aligned(16))) __bf16 sX[128 * LPAD];
    __shared__ __attribute__((aligned(16))) __bf16 sA[16 * LPAD];
    const int st = blockIdx.x, j = blockIdx.y, b = blockIdx.z;
    const int tid = threadIdx.x, wv = tid >> 5, ln = tid & 31;
    const int e = top2i[b * 2 + j];
    const float wgt = top2w[b * 2 + j];
    const int s0 = st * 128;

    const int m   = ln & 15;
    const int kb  = (ln < 16) ? 0 : 8;    // A-operand K split
    const int kb2 = (ln < 16) ? 0 : 16;   // unused; B operand also uses kb
    v8f acc = {};

    for (int k0 = 0; k0 < Dn; k0 += 32) {
        __syncthreads();
        // stage X chunk [128 s][32 k] f32 -> bf16 LDS
#pragma unroll
        for (int it = 0; it < 4; ++it) {
            const int idx = tid + it * 256;         // 0..1023
            const int row = idx >> 3, q = idx & 7;  // q = float4 within the 32 cols
            const float4 v = *(const float4*)(X + ((size_t)b * Sn + s0 + row) * Dn + k0 + q * 4);
            __bf16* d = &sX[row * LPAD + q * 4];
            d[0] = (__bf16)v.x; d[1] = (__bf16)v.y; d[2] = (__bf16)v.z; d[3] = (__bf16)v.w;
        }
        // stage A_e chunk [16 r][32 k]
        if (tid < 128) {
            const int row = tid >> 3, q = tid & 7;
            const float4 v = *(const float4*)(loraA + ((size_t)e * Rn + row) * Dn + k0 + q * 4);
            __bf16* d = &sA[row * LPAD + q * 4];
            d[0] = (__bf16)v.x; d[1] = (__bf16)v.y; d[2] = (__bf16)v.z; d[3] = (__bf16)v.w;
        }
        if (k0 + 32 < Dn)
            __builtin_prefetch(X + ((size_t)b * Sn + s0 + (tid >> 3)) * Dn + k0 + 32, 0, 1);
        __syncthreads();

        const v8bf alo = *(const v8bf*)&sX[(wv * 16 + m) * LPAD + kb];
        const v8bf ahi = *(const v8bf*)&sX[(wv * 16 + m) * LPAD + 16 + kb];
        const v8bf blo = *(const v8bf*)&sA[m * LPAD + kb];
        const v8bf bhi = *(const v8bf*)&sA[m * LPAD + 16 + kb];
        acc = __builtin_amdgcn_wmma_f32_16x16x32_bf16(
            false, make16(alo, ahi), false, make16(blo, bhi),
            (short)0, acc, false, false);
    }
    // store weighted bf16 mid; C layout: vgpr i -> M = i (+8 for lanes>=16), N = ln&15
    const int n = ln & 15;
    const int mo = (ln < 16) ? 0 : 8;
#pragma unroll
    for (int i = 0; i < 8; ++i) {
        const int s = s0 + wv * 16 + i + mo;
        mid[(((size_t)b * 2 + j) * Sn + s) * Rn + n] = (__bf16)(acc[i] * wgt);
    }
}

// ---------------------------------------------------------------------------
// Kernel 4: out[b] = midcat[S,32] @ Bcat[32,D]  — one K=32 WMMA per 16x16 tile
// grid (D/128, S/128, B), block 256.
// ---------------------------------------------------------------------------
__global__ void out_kernel(const __bf16* __restrict__ mid,
                           const float* __restrict__ loraB,
                           const int* __restrict__ top2i,
                           float* __restrict__ out) {
    __shared__ __attribute__((aligned(16))) __bf16 sM[128 * LPAD];  // [s][k]
    __shared__ __attribute__((aligned(16))) __bf16 sB[128 * LPAD];  // [d][k] (transposed)
    const int dt = blockIdx.x, st = blockIdx.y, b = blockIdx.z;
    const int tid = threadIdx.x, wv = tid >> 5, ln = tid & 31;
    const int e0 = top2i[b * 2], e1 = top2i[b * 2 + 1];

    {   // stage mid tile: 128 rows x (2 experts x 16) bf16
        const int row = tid >> 1, j = tid & 1;
        const __bf16* src = mid + (((size_t)b * 2 + j) * Sn + st * 128 + row) * Rn;
        __bf16* d = &sM[row * LPAD + j * 16];
        *(float4*)(d)     = *(const float4*)(src);
        *(float4*)(d + 8) = *(const float4*)(src + 8);
    }
    {   // stage Bcat tile: 128 d-rows x 32 k, element (d, j*16+r) = loraB[e_j][d][r]
        const int d = tid >> 1, jj = tid & 1;
        const int e = jj ? e1 : e0;
        const float* src = loraB + ((size_t)e * Dn + dt * 128 + d) * Rn;
        __bf16* dst = &sB[d * LPAD + jj * 16];
#pragma unroll
        for (int q = 0; q < 4; ++q) {
            const float4 v = *(const float4*)(src + q * 4);
            dst[q * 4 + 0] = (__bf16)v.x; dst[q * 4 + 1] = (__bf16)v.y;
            dst[q * 4 + 2] = (__bf16)v.z; dst[q * 4 + 3] = (__bf16)v.w;
        }
    }
    __syncthreads();

    const int m   = ln & 15;
    const int kb  = (ln < 16) ? 0 : 8;
    const int kb2 = (ln < 16) ? 0 : 16;
    const v8bf alo = *(const v8bf*)&sM[(wv * 16 + m) * LPAD + kb];
    const v8bf ahi = *(const v8bf*)&sM[(wv * 16 + m) * LPAD + 16 + kb];
    const v16bf av = make16(alo, ahi);

    const int n  = ln & 15;
    const int mo = (ln < 16) ? 0 : 8;
    float* obase = out + ((size_t)b * Sn + st * 128 + wv * 16) * Dn + dt * 128;

    for (int dsub = 0; dsub < 8; ++dsub) {
        const v8bf blo = *(const v8bf*)&sB[(dsub * 16 + n) * LPAD + kb];
        const v8bf bhi = *(const v8bf*)&sB[(dsub * 16 + n) * LPAD + 16 + kb];
        v8f c = {};
        c = __builtin_amdgcn_wmma_f32_16x16x32_bf16(
            false, av, false, make16(blo, bhi), (short)0, c, false, false);
        float* op = obase + dsub * 16;
#pragma unroll
        for (int i = 0; i < 8; ++i)
            op[(size_t)(i + mo) * Dn + n] = c[i];
    }
}

// ---------------------------------------------------------------------------
extern "C" void kernel_launch(void* const* d_in, const int* in_sizes, int n_in,
                              void* d_out, int out_size, void* d_ws, size_t ws_size,
                              hipStream_t stream) {
    const float* X      = (const float*)d_in[0];   // [B,S,D]
    const float* gate_w = (const float*)d_in[1];   // [E,D]
    const float* tb     = (const float*)d_in[2];   // [4,E]
    const float* mb     = (const float*)d_in[3];   // [2,E]
    const float* loraA  = (const float*)d_in[4];   // [E,R,D]
    const float* loraB  = (const float*)d_in[5];   // [E,D,R]
    const int*   t_id   = (const int*)d_in[6];
    const int*   m_id   = (const int*)d_in[7];
    float*       out    = (float*)d_out;           // [B,S,D]

    // workspace layout
    char* ws = (char*)d_ws;
    float*  partial = (float*)ws;                                   // B*PC*D f32 = 2 MB
    size_t  off = (size_t)Bn * PC * Dn * sizeof(float);
    int*    top2i = (int*)(ws + off);                               // 16 ints
    float*  top2w = (float*)(ws + off + 64);                        // 16 floats
    __bf16* mid   = (__bf16*)(ws + off + 256);                      // B*2*S*R bf16 = 1 MB

    pool_kernel<<<dim3(PC, Bn), 256, 0, stream>>>(X, partial);
    router_kernel<<<1, 256, 0, stream>>>(partial, gate_w, tb, mb, t_id, m_id, top2i, top2w);
    mid_kernel<<<dim3(Sn / 128, 2, Bn), 256, 0, stream>>>(X, loraA, top2i, top2w, mid);
    out_kernel<<<dim3(Dn / 128, Sn / 128, Bn), 256, 0, stream>>>(mid, loraB, top2i, out);
}
